// ChainCRF_85993835200882
// MI455X (gfx1250) — matrix operand order repaired and checked by reading, not compiled, not run on hardware
//
#include <hip/hip_runtime.h>

#define N_TAGS   128
#define ROOT_TAG 126
#define END_TAG  127
#define T_LEN    256
#define BSZ      256
#define ROWS     16      // batch rows per workgroup

#define LOG2E 1.4426950408889634f
#define LN2   0.6931471805599453f

typedef __attribute__((ext_vector_type(16))) _Float16     v16h;
typedef __attribute__((ext_vector_type(8)))  float        v8f;
typedef __attribute__((ext_vector_type(4)))  unsigned int v4u;

union V16H { v16h v; _Float16 e[16]; v4u u4[2]; };
union V8F  { v8f  v; float    f[8]; };

__global__ void crf_init_ws(float* __restrict__ ws) {
    ws[0] = 0.0f;   // sum of partitions
    ws[1] = 0.0f;   // sum of gold-path scores
}

// Forward recursion: one workgroup owns ROWS batch rows for all T steps.
// 8 waves; each wave owns a 16-wide state-column tile and keeps its
// exp(transition) B-operands resident in VGPRs for the entire scan.
__global__ __launch_bounds__(256) void crf_forward(
    const float* __restrict__ feats,   // [BSZ][T][N_TAGS]
    const int*   __restrict__ mask,    // [BSZ][T]
    const float* __restrict__ lt,      // [N_TAGS][N_TAGS]
    float*       __restrict__ accum)   // ws
{
    __shared__ __align__(16) float    fv[ROWS][N_TAGS];   // forward variables
    __shared__ __align__(16) _Float16 P [ROWS][N_TAGS];   // exp(fv - rowmax)
    __shared__ __align__(16) float    mk_f[ROWS][T_LEN];  // mask slab as 0/1 f32
    __shared__ float rowmax[ROWS];

    const int tid  = threadIdx.x;
    const int wave = tid >> 5;
    const int lane = tid & 31;
    const int hi   = lane >> 4;      // half-wave select
    const int lm   = lane & 15;      // row (A) / column-in-tile (B,C,D)
    const int b0   = blockIdx.x * ROWS;
    const int n    = wave * 16 + lm; // this lane's output state column

    // ---- Resident B operands: E[k][n] = exp(lt[k][n]) --------------------
    // B k order chosen here (contiguous, A matches): lanes 0-15 hold K=h, lanes 16-31 hold K=16+h.
    V16H bmat[4];
    #pragma unroll
    for (int q = 0; q < 4; ++q) {
        #pragma unroll
        for (int h = 0; h < 16; ++h) {
            const int k = q * 32 + 16 * hi + h;
            bmat[q].e[h] = (_Float16)__builtin_amdgcn_exp2f(lt[k * N_TAGS + n] * LOG2E);
        }
    }

    // ---- preload this WG's mask slab as float 0/1 (contiguous rows) ------
    {
        const int4* __restrict__ src = (const int4*)(mask + (size_t)b0 * T_LEN);
        float4* dst = (float4*)&mk_f[0][0];
        for (int i = tid; i < ROWS * T_LEN / 4; i += 256) {
            const int4 v = src[i];
            dst[i] = make_float4((float)v.x, (float)v.y, (float)v.z, (float)v.w);
        }
    }

    // ---- init: fv = lt[ROOT,:] + feats[:,0,:] ----------------------------
    for (int i = tid; i < ROWS * N_TAGS; i += 256) {
        const int r = i >> 7, j = i & 127;
        fv[r][j] = lt[ROOT_TAG * N_TAGS + j]
                 + feats[((size_t)(b0 + r) * T_LEN) * N_TAGS + j];
    }
    __syncthreads();

    // Private mirror of the 8 fv entries this lane owns (rows r8+8*hi, col n)
    float fvreg[8];
    #pragma unroll
    for (int r8 = 0; r8 < 8; ++r8) fvreg[r8] = fv[r8 + 8 * hi][n];

    const int rr = tid >> 4;   // row this thread helps reduce (16 thr/row)
    const int cc = tid & 15;
    const float* __restrict__ featp =
        feats + ((size_t)(b0 + 8 * hi) * T_LEN) * N_TAGS + n;

    for (int t = 1; t < T_LEN; ++t) {
        // -- row max via in-wave shuffle reduce (16 lanes share a row) ----
        {
            const int c0 = cc * 8;
            float m = fv[rr][c0];
            #pragma unroll
            for (int u = 1; u < 8; ++u) m = fmaxf(m, fv[rr][c0 + u]);
            m = fmaxf(m, __shfl_xor(m, 1));
            m = fmaxf(m, __shfl_xor(m, 2));
            m = fmaxf(m, __shfl_xor(m, 4));
            m = fmaxf(m, __shfl_xor(m, 8));
            if (cc == 0) rowmax[rr] = m;
            // -- P = exp(fv - m), f16 (raw v_exp_f32 via exp2) ------------
            #pragma unroll
            for (int u = 0; u < 8; ++u)
                P[rr][c0 + u] = (_Float16)__builtin_amdgcn_exp2f(
                                    (fv[rr][c0 + u] - m) * LOG2E);
        }
        __syncthreads();

        // -- issue feat loads early to overlap with the WMMA chain --------
        float fvals[8];
        #pragma unroll
        for (int r8 = 0; r8 < 8; ++r8)
            fvals[r8] = featp[(size_t)r8 * T_LEN * N_TAGS + (size_t)t * N_TAGS];

        // -- preload all four A operands as one LDS clause ----------------
        // A 16x32 f16 layout: halves 0-15 -> K=q*32+16*hi+h
        // (matches B). Two b128 LDS loads each.
        V16H a[4];
        #pragma unroll
        for (int q = 0; q < 4; ++q) {
            a[q].u4[0] = *(const v4u*)&P[lm][q * 32 + 16 * hi];
            a[q].u4[1] = *(const v4u*)&P[lm][q * 32 + 16 * hi + 8];
        }

        // -- logsumexp core: back-to-back accumulating WMMAs --------------
        v8f acc = {};
        #pragma unroll
        for (int q = 0; q < 4; ++q)
            acc = __builtin_amdgcn_wmma_f32_16x16x32_f16(
                      false, a[q].v, false, bmat[q].v, (short)0, acc, false, false);

        // -- nxt = log(C) + M + feat ; arithmetic (unbranchable) blend ----
        V8F c; c.v = acc;
        #pragma unroll
        for (int r8 = 0; r8 < 8; ++r8) {
            const int m = r8 + 8 * hi;                  // C/D row layout
            const float nxt =
                __builtin_amdgcn_logf(fmaxf(c.f[r8], 1e-30f)) * LN2
                + rowmax[m] + fvals[r8];
            const float mf = mk_f[m][t];                // 0.0 or 1.0
            fvreg[r8] = fmaf(mf, nxt - fvreg[r8], fvreg[r8]);
            fv[m][n]  = fvreg[r8];
        }
        __syncthreads();
    }

    // ---- partition_b = logsumexp_j(fv[b,j] + lt[j,END]) ------------------
    if (tid < ROWS) {
        float m = -3.4e38f;
        for (int j = 0; j < N_TAGS; ++j)
            m = fmaxf(m, fv[tid][j] + lt[j * N_TAGS + END_TAG]);
        float s = 0.0f;
        for (int j = 0; j < N_TAGS; ++j)
            s += __builtin_amdgcn_exp2f((fv[tid][j] + lt[j * N_TAGS + END_TAG] - m) * LOG2E);
        atomicAdd(&accum[0], m + __builtin_amdgcn_logf(s) * LN2);
    }
}

// Gold-path score: one thread per batch row (cheap gathers, branchless).
__global__ __launch_bounds__(256) void crf_score(
    const float* __restrict__ feats, const int* __restrict__ tags,
    const int* __restrict__ mask, const float* __restrict__ lt,
    float* __restrict__ accum)
{
    const int b = blockIdx.x * blockDim.x + threadIdx.x;
    if (b >= BSZ) return;
    const int*   tg = tags + b * T_LEN;
    const int*   mk = mask + b * T_LEN;
    const float* fb = feats + (size_t)b * T_LEN * N_TAGS;

    float s = lt[ROOT_TAG * N_TAGS + tg[0]];
    int cnt = 0;
    for (int t = 0; t < T_LEN; ++t) cnt += mk[t];
    for (int t = 1; t < T_LEN; ++t)
        s += lt[tg[t - 1] * N_TAGS + tg[t]] * (float)mk[t];
    for (int t = 0; t < T_LEN - 1; ++t)
        s += fb[(size_t)t * N_TAGS + tg[t]] * (float)mk[t];
    const int ltag = tg[cnt - 1];
    s += lt[ltag * N_TAGS + END_TAG];
    s += fb[(size_t)(T_LEN - 1) * N_TAGS + ltag] * (float)mk[T_LEN - 1];
    atomicAdd(&accum[1], s);
}

__global__ void crf_finalize(const float* __restrict__ ws, float* __restrict__ out) {
    out[0] = (ws[0] - ws[1]) * (1.0f / (float)BSZ);
}

extern "C" void kernel_launch(void* const* d_in, const int* in_sizes, int n_in,
                              void* d_out, int out_size, void* d_ws, size_t ws_size,
                              hipStream_t stream) {
    (void)in_sizes; (void)n_in; (void)out_size; (void)ws_size;
    const float* feats = (const float*)d_in[0];
    const int*   tags  = (const int*)d_in[1];
    const int*   mask  = (const int*)d_in[2];
    const float* lt    = (const float*)d_in[3];
    float* out = (float*)d_out;
    float* ws  = (float*)d_ws;

    crf_init_ws<<<1, 1, 0, stream>>>(ws);
    crf_forward<<<BSZ / ROWS, 256, 0, stream>>>(feats, mask, lt, ws);
    crf_score<<<1, 256, 0, stream>>>(feats, tags, mask, lt, ws);
    crf_finalize<<<1, 1, 0, stream>>>(ws, out);
}
